// NTXentLoss_85641647882654
// MI455X (gfx1250) — compile-verified, matrix operand order checked by reading, not run on hardware
//
#include <hip/hip_runtime.h>
#include <math.h>

// ---------------------------------------------------------------------------
// NT-Xent loss, MI455X (gfx1250, wave32, WMMA + async LDS staging).
//
//   loss_i = log( sum_{j != i} exp(sim[i,j]/T) ) - sim[i, pos(i)]/T
//   sim    = zn @ zn^T,  zn = z / max(||z||, eps)
//
// sim/T bounded in [-2,2] -> plain sum-of-exp (no max-shift needed).
// Gram matrix streamed tile-by-tile through v_wmma_f32_16x16x32_f16.
//
// Data path: each block stages one 16x256 f16 column tile (8 KB) in LDS via
// global_load_async_to_lds_b128 (ASYNCcnt), double-buffered one tile ahead;
// the 4 waves (4 row blocks) compute from LDS. Hot loop is branch-free:
// diagonal / positive tiles are recomputed after the sweep (WMMA is
// deterministic) and corrections applied under uniform branches.
// ---------------------------------------------------------------------------

typedef __attribute__((ext_vector_type(16))) _Float16 v16h;
typedef __attribute__((ext_vector_type(8)))  _Float16 v8h;
typedef __attribute__((ext_vector_type(8)))  float    v8f;
typedef __attribute__((ext_vector_type(4)))  int      v4i;

// Pointer types matching the async-copy builtin's parameters:
// param0: global (AS1) int4*, param1: LDS (AS3) int4* (per hipcc diagnostic).
typedef __attribute__((address_space(1))) v4i* as1_v4i_p;
typedef __attribute__((address_space(3))) v4i* as3_v4i_p;

#define B_    2048
#define N_    4096      // 2*B
#define D_    256
#define INV_T 2.0f      // 1 / 0.5
#define EPS_  1e-8f

#define WAVES_PER_BLOCK 4
#define ROW_BLOCKS (N_ / 16)        // 256
#define COL_SPLIT  4                // column-sweep split for occupancy
#define SEG_COLS   (N_ / COL_SPLIT) // 1024 cols per segment
#define SEG_TILES  (SEG_COLS / 16)  // 64 tiles per segment
#define TOTAL_BLOCKS (ROW_BLOCKS / WAVES_PER_BLOCK * COL_SPLIT)  // 256

// ---------------- async global -> LDS copy (16B per lane) ------------------
__device__ __forceinline__ void async_copy16(const _Float16* g, _Float16* l) {
#if __has_builtin(__builtin_amdgcn_global_load_async_to_lds_b128)
  __builtin_amdgcn_global_load_async_to_lds_b128(
      (as1_v4i_p)g, (as3_v4i_p)l, 0, 0);
#else
  *(v8h*)l = *(const v8h*)g;   // sync fallback: global_load + ds_store
#endif
}

__device__ __forceinline__ void wait_async() {
#if __has_builtin(__builtin_amdgcn_s_wait_asynccnt)
  __builtin_amdgcn_s_wait_asynccnt(0);
#elif __has_builtin(__builtin_amdgcn_global_load_async_to_lds_b128)
  asm volatile("s_wait_asynccnt 0x0" ::: "memory");
#endif
}

// ---------------------------------------------------------------------------
// Kernel 1: row-normalize [z_i; z_j] -> f16 zn in workspace.
// ---------------------------------------------------------------------------
__global__ void ntxent_normalize(const float* __restrict__ zi,
                                 const float* __restrict__ zj,
                                 _Float16* __restrict__ zh) {
  const int row  = blockIdx.x * (blockDim.x >> 5) + (threadIdx.x >> 5);
  const int lane = threadIdx.x & 31;
  if (row >= N_) return;

  const float* src = (row < B_) ? (zi + (size_t)row * D_)
                                : (zj + (size_t)(row - B_) * D_);
  float4 v0 = *(const float4*)(src + lane * 8);
  float4 v1 = *(const float4*)(src + lane * 8 + 4);

  float ss = v0.x * v0.x + v0.y * v0.y + v0.z * v0.z + v0.w * v0.w +
             v1.x * v1.x + v1.y * v1.y + v1.z * v1.z + v1.w * v1.w;
#pragma unroll
  for (int m = 16; m >= 1; m >>= 1) ss += __shfl_xor(ss, m, 32);

  const float scale = 1.0f / fmaxf(sqrtf(ss), EPS_);

  v8h h;
  h[0] = (_Float16)(v0.x * scale);  h[1] = (_Float16)(v0.y * scale);
  h[2] = (_Float16)(v0.z * scale);  h[3] = (_Float16)(v0.w * scale);
  h[4] = (_Float16)(v1.x * scale);  h[5] = (_Float16)(v1.y * scale);
  h[6] = (_Float16)(v1.z * scale);  h[7] = (_Float16)(v1.w * scale);
  *(v8h*)(zh + (size_t)row * D_ + lane * 8) = h;
}

// ------------- fragment loaders (16-bit 16x32 A / B^T layout) --------------
// lane g=lane&15 -> row/col (base+g); h=lane>>4 selects K {h*8.., 16+h*8..}.
__device__ __forceinline__ v16h load_frag_g(const _Float16* __restrict__ zh,
                                            int rc0, int k0, int lane) {
  const int g = lane & 15, h = lane >> 4;
  const _Float16* p = zh + (size_t)(rc0 + g) * D_ + k0 + h * 8;
  v8h lo = *(const v8h*)p;
  v8h hi = *(const v8h*)(p + 16);
  return __builtin_shufflevector(lo, hi, 0, 1, 2, 3, 4, 5, 6, 7,
                                         8, 9, 10, 11, 12, 13, 14, 15);
}

__device__ __forceinline__ v16h load_frag_lds(const _Float16* base,
                                              int k0, int lane) {
  const int g = lane & 15, h = lane >> 4;
  const _Float16* p = base + g * D_ + k0 + h * 8;
  v8h lo = *(const v8h*)p;          // ds_load_b128
  v8h hi = *(const v8h*)(p + 16);   // ds_load_b128
  return __builtin_shufflevector(lo, hi, 0, 1, 2, 3, 4, 5, 6, 7,
                                         8, 9, 10, 11, 12, 13, 14, 15);
}

// One 16x16 Gram tile against preloaded A fragments, B from global.
__device__ __forceinline__ v8f tile_gram_g(const _Float16* __restrict__ zh,
                                           const v16h* a, int c0, int lane) {
  v8f c = {};
#pragma unroll
  for (int kc = 0; kc < 8; ++kc) {
    v16h b = load_frag_g(zh, c0, kc * 32, lane);
    c = __builtin_amdgcn_wmma_f32_16x16x32_f16(false, a[kc], false, b,
                                               (short)0, c, false, false);
  }
  return c;
}

// ---------------------------------------------------------------------------
// Kernel 2: block -> column segment (blockIdx & 3) + 4 row blocks (1/wave).
// ---------------------------------------------------------------------------
__global__ void __launch_bounds__(WAVES_PER_BLOCK * 32)
ntxent_main(const _Float16* __restrict__ zh,
            float* __restrict__ Srow,    // [COL_SPLIT][N_]
            float* __restrict__ Prow) {  // [COL_SPLIT][N_]
  __shared__ _Float16 lbuf[2][16 * D_];  // 2 x 8KB double buffer

  const int tid  = threadIdx.x;
  const int lane = tid & 31;
  const int wave = tid >> 5;
  const int seg  = blockIdx.x & (COL_SPLIT - 1);
  const int m0   = (((int)blockIdx.x >> 2) * WAVES_PER_BLOCK + wave) * 16;
  const int cseg = seg * SEG_COLS;
  const int c0p  = (m0 + B_) & (N_ - 1);   // positives' tile base

  // Preload this wave's A fragments (16 rows x K=256).
  v16h a[8];
#pragma unroll
  for (int kc = 0; kc < 8; ++kc) a[kc] = load_frag_g(zh, m0, kc * 32, lane);

  float S[8];
#pragma unroll
  for (int r = 0; r < 8; ++r) S[r] = 0.0f;

  const int g = lane & 15;   // C/D layout: col within tile = lane % 16
  const int h = lane >> 4;   // row-half: M = r + 8*h
  const int coff = tid * 32; // this thread's 64B slice of the 8KB tile

  // Prologue: stage tile 0 into buffer 0.
#pragma unroll
  for (int j = 0; j < 4; ++j)
    async_copy16(zh + (size_t)cseg * D_ + coff + j * 8, &lbuf[0][coff + j * 8]);
  wait_async();
  __syncthreads();

  for (int t = 0; t < SEG_TILES; ++t) {
    const int cb = t & 1;
    // Issue async copy of the next tile (wraps within segment; the wrap
    // iteration is a harmless re-copy) into the other buffer.
    const int nc0 = cseg + ((t + 1) & (SEG_TILES - 1)) * 16;
    _Float16* nbase = &lbuf[cb ^ 1][0];
#pragma unroll
    for (int j = 0; j < 4; ++j)
      async_copy16(zh + (size_t)nc0 * D_ + coff + j * 8, nbase + coff + j * 8);

    // Compute current tile from LDS: 16 ds_load_b128 + 8 chained WMMAs.
    const _Float16* cbase = &lbuf[cb][0];
    v16h b[8];
#pragma unroll
    for (int kc = 0; kc < 8; ++kc) b[kc] = load_frag_lds(cbase, kc * 32, lane);

    v8f c = {};
#pragma unroll
    for (int kc = 0; kc < 8; ++kc)
      c = __builtin_amdgcn_wmma_f32_16x16x32_f16(false, a[kc], false, b[kc],
                                                 (short)0, c, false, false);

    // Branch-free hot path: every element -> mul + v_exp_f32 + add.
#pragma unroll
    for (int r = 0; r < 8; ++r) S[r] += __expf(c[r] * INV_T);

    wait_async();
    __syncthreads();
  }

  // ---- corrections (uniform per-wave branches; EXEC stays all-ones) ----
  float P[8];
#pragma unroll
  for (int r = 0; r < 8; ++r) P[r] = 0.0f;

  if (m0 >= cseg && m0 < cseg + SEG_COLS) {      // diagonal tile in my seg
    v8f c = tile_gram_g(zh, a, m0, lane);        // identical to in-loop tile
#pragma unroll
    for (int r = 0; r < 8; ++r)
      S[r] -= (g == r + 8 * h) ? __expf(c[r] * INV_T) : 0.0f;
  }
  if (c0p >= cseg && c0p < cseg + SEG_COLS) {    // positives' tile in my seg
    v8f c = tile_gram_g(zh, a, c0p, lane);
#pragma unroll
    for (int r = 0; r < 8; ++r)
      P[r] = (g == r + 8 * h) ? c[r] * INV_T : P[r];
  }

  // Row-wise reduction: row (m0+r+8h) lives in vgpr slot r across 16 lanes.
#pragma unroll
  for (int r = 0; r < 8; ++r) {
    float s = S[r], p = P[r];
#pragma unroll
    for (int m = 8; m >= 1; m >>= 1) {
      s += __shfl_xor(s, m, 16);
      p += __shfl_xor(p, m, 16);
    }
    if ((lane & 15) == 0) {
      const int row = m0 + r + h * 8;
      Srow[seg * N_ + row] = s;
      Prow[seg * N_ + row] = p;
    }
  }
}

// ---------------------------------------------------------------------------
// Kernel 3: single 1024-thread block; loss = log(sum_seg S) - P per row.
// ---------------------------------------------------------------------------
__global__ void ntxent_reduce(const float* __restrict__ Srow,
                              const float* __restrict__ Prow,
                              float* __restrict__ out) {
  __shared__ float wsum[32];
  const int tid  = threadIdx.x;
  const int lane = tid & 31;
  const int wid  = tid >> 5;

  float acc = 0.0f;
  for (int row = tid; row < N_; row += 1024) {
    float s = 0.0f, p = 0.0f;
#pragma unroll
    for (int seg = 0; seg < COL_SPLIT; ++seg) {
      s += Srow[seg * N_ + row];
      p += Prow[seg * N_ + row];
    }
    acc += logf(s) - p;
  }
#pragma unroll
  for (int m = 16; m >= 1; m >>= 1) acc += __shfl_xor(acc, m, 32);
  if (lane == 0) wsum[wid] = acc;
  __syncthreads();
  if (wid == 0) {
    float v = wsum[lane];
#pragma unroll
    for (int m = 16; m >= 1; m >>= 1) v += __shfl_xor(v, m, 32);
    if (lane == 0) out[0] = v * (1.0f / (float)N_);
  }
}

// ---------------------------------------------------------------------------
extern "C" void kernel_launch(void* const* d_in, const int* in_sizes, int n_in,
                              void* d_out, int out_size, void* d_ws, size_t ws_size,
                              hipStream_t stream) {
  (void)in_sizes; (void)n_in; (void)out_size; (void)ws_size;
  const float* zi = (const float*)d_in[0];
  const float* zj = (const float*)d_in[1];
  float* out = (float*)d_out;

  _Float16* zh = (_Float16*)d_ws;                                   // 2 MB
  char* p = (char*)d_ws + (size_t)N_ * D_ * sizeof(_Float16);
  float* Srow = (float*)p;
  float* Prow = (float*)(p + (size_t)COL_SPLIT * N_ * sizeof(float));

  ntxent_normalize<<<N_ / 8, 256, 0, stream>>>(zi, zj, zh);
  ntxent_main<<<TOTAL_BLOCKS, WAVES_PER_BLOCK * 32, 0, stream>>>(zh, Srow, Prow);
  ntxent_reduce<<<1, 1024, 0, stream>>>(Srow, Prow, out);
}
